// Router_27238682591310
// MI455X (gfx1250) — matrix operand order repaired and checked by reading, not compiled, not run on hardware
//
#include <hip/hip_runtime.h>
#include <hip/hip_bf16.h>
#include <math.h>

typedef _Float16 v16h __attribute__((ext_vector_type(16)));
typedef _Float16 v8h  __attribute__((ext_vector_type(8)));
typedef float    v8f  __attribute__((ext_vector_type(8)));
typedef float    v4f  __attribute__((ext_vector_type(4)));   // clang vector (works with nontemporal builtins)

#define D_MODEL 2048
#define NEXP    8
#define NTOK    16384
#define SPR_T   256

// ---------- helpers: build WMMA fragments (CDNA5 wave32 16x16x32 f16 layouts) ----------
// A 16x32 f16: lane l (m=l&15, half=l>>4): elems 0..7 = K(k0+half*8+e), elems 8..15 = K(k0+16+half*8+e)
// B 32x16 f16: lane l (n=l&15, half=l>>4): elems 0..7 = K(k0+half*8+e), elems 8..15 = K(k0+16+half*8+e)

__device__ __forceinline__ v16h loadA_f32_nt(const float* row, int k0, int half) {
  const float* lo = row + k0 + half * 8;
  const float* hi = lo + 16;
  v4f u0 = __builtin_nontemporal_load((const v4f*)lo);
  v4f u1 = __builtin_nontemporal_load((const v4f*)(lo + 4));
  v4f u2 = __builtin_nontemporal_load((const v4f*)hi);
  v4f u3 = __builtin_nontemporal_load((const v4f*)(hi + 4));
  v16h a;
#pragma unroll
  for (int e = 0; e < 4; ++e) {
    a[e]      = (_Float16)u0[e];
    a[4 + e]  = (_Float16)u1[e];
    a[8 + e]  = (_Float16)u2[e];
    a[12 + e] = (_Float16)u3[e];
  }
  return a;
}

__device__ __forceinline__ v16h loadA_h16(const _Float16* row, int k0, int half) {
  v8h a0 = *(const v8h*)(row + k0 + half * 8);
  v8h a1 = *(const v8h*)(row + k0 + 16 + half * 8);
  v16h a;
#pragma unroll
  for (int e = 0; e < 8; ++e) { a[e] = a0[e]; a[8 + e] = a1[e]; }
  return a;
}

__device__ __forceinline__ v16h loadB_h16(const _Float16* row, int k0, int half) {
  v8h b0 = *(const v8h*)(row + k0 + half * 8);
  v8h b1 = *(const v8h*)(row + k0 + 16 + half * 8);
  v16h b;
#pragma unroll
  for (int e = 0; e < 8; ++e) { b[e] = b0[e]; b[8 + e] = b1[e]; }
  return b;
}

// ---------- kernel 0: zero workspace accumulators ----------
__global__ void k_init(float* __restrict__ acc) {
  if (threadIdx.x < 17) acc[threadIdx.x] = 0.f;   // [0..7]=counts, [8..15]=prob sums, [16]=spr sq-sum
}

// ---------- kernel 1: gate GEMM (WMMA) + softmax + top-2 + per-expert partials ----------
__global__ __launch_bounds__(256) void k_router_gemm(
    const float* __restrict__ x, const float* __restrict__ W,
    float* __restrict__ out, float* __restrict__ acc)
{
  // 9 rows: 8 real experts + 1 shared zero row for the padded WMMA columns (branch-free B loads)
  __shared__ __align__(16) _Float16 wlds[(NEXP + 1) * D_MODEL];  // 36 KB
  __shared__ float ltile[8][16][NEXP];                           // per-wave logits tile
  __shared__ float psum_s[NEXP];
  __shared__ float cnt_s[NEXP];

  const int tid = threadIdx.x;

  // stage W -> LDS (f16); zero the padding row and block accumulators
  const v4f* W4 = (const v4f*)W;
  for (int i = tid; i < NEXP * D_MODEL / 4; i += 256) {
    v4f w = W4[i];
    wlds[4 * i + 0] = (_Float16)w[0];
    wlds[4 * i + 1] = (_Float16)w[1];
    wlds[4 * i + 2] = (_Float16)w[2];
    wlds[4 * i + 3] = (_Float16)w[3];
  }
  for (int i = tid; i < D_MODEL; i += 256) wlds[NEXP * D_MODEL + i] = (_Float16)0.f;
  if (tid < NEXP) { psum_s[tid] = 0.f; cnt_s[tid] = 0.f; }
  __syncthreads();

  const int wave = tid >> 5;
  const int lane = tid & 31;
  const int half = lane >> 4;
  const int m    = lane & 15;
  const int t0   = (blockIdx.x * 8 + wave) * 16;

  const float*    xrow = x + (size_t)(t0 + m) * D_MODEL;
  const _Float16* brow = wlds + (size_t)(m < NEXP ? m : NEXP) * D_MODEL;  // n>=8 -> zero row

  v8f c = {};
  for (int k0 = 0; k0 < D_MODEL; k0 += 32) {
    v16h a = loadA_f32_nt(xrow, k0, half);
    v16h b = loadB_h16(brow, k0, half);
    c = __builtin_amdgcn_wmma_f32_16x16x32_f16(false, a, false, b, (short)0, c, false, false);
  }

  // scatter C (M = half*8 + r, N = m) to LDS; only 8 real expert columns
  if (m < NEXP) {
#pragma unroll
    for (int r = 0; r < 8; ++r) ltile[wave][half * 8 + r][m] = c[r];
  }
  __syncthreads();

  // lanes 0..15: one token each -> softmax over 8 experts, top-2, outputs + partials
  if (lane < 16) {
    const int tok = t0 + lane;
    float p[NEXP];
    float mx = -1e30f;
#pragma unroll
    for (int e = 0; e < NEXP; ++e) { p[e] = ltile[wave][lane][e]; mx = fmaxf(mx, p[e]); }
    float s = 0.f;
#pragma unroll
    for (int e = 0; e < NEXP; ++e) { p[e] = expf(p[e] - mx); s += p[e]; }
    const float inv = 1.f / s;
#pragma unroll
    for (int e = 0; e < NEXP; ++e) {
      p[e] *= inv;
      out[(size_t)tok * NEXP + e] = p[e];
      atomicAdd(&psum_s[e], p[e]);
    }
    int i1 = 0; float p1 = p[0];
#pragma unroll
    for (int e = 1; e < NEXP; ++e) if (p[e] > p1) { p1 = p[e]; i1 = e; }
    int i2 = (i1 == 0) ? 1 : 0; float p2 = p[i2];
#pragma unroll
    for (int e = 0; e < NEXP; ++e) if (e != i1 && p[e] > p2) { p2 = p[e]; i2 = e; }
    const float d = 1.f / (p1 + p2);
    float* idx_out = out + (size_t)NTOK * NEXP;
    float* tp_out  = idx_out + (size_t)NTOK * 2;
    idx_out[tok * 2 + 0] = (float)i1;
    idx_out[tok * 2 + 1] = (float)i2;
    tp_out[tok * 2 + 0]  = p1 * d;
    tp_out[tok * 2 + 1]  = p2 * d;
    atomicAdd(&cnt_s[i1], 1.f);
    atomicAdd(&cnt_s[i2], 1.f);
  }
  __syncthreads();
  if (tid < NEXP) {
    atomicAdd(&acc[tid],     cnt_s[tid]);
    atomicAdd(&acc[8 + tid], psum_s[tid]);
  }
}

// ---------- kernel 2a: gather + L2-normalize sampled tokens (f16) and router rows ----------
__global__ __launch_bounds__(256) void k_spr_gather(
    const float* __restrict__ x, const int* __restrict__ spr_idx,
    const float* __restrict__ rp,                  // router_probs (= d_out head)
    _Float16* __restrict__ xn, _Float16* __restrict__ rn)
{
  __shared__ float red[256];
  const int tid = threadIdx.x;
  const int b   = blockIdx.x;
  const int tok = spr_idx[b];
  const float* xr = x + (size_t)tok * D_MODEL;

  float s = 0.f;
  for (int i = tid; i < D_MODEL; i += 256) {
    float v = __builtin_nontemporal_load(xr + i);
    s += v * v;
  }
  red[tid] = s; __syncthreads();
  for (int st = 128; st > 0; st >>= 1) { if (tid < st) red[tid] += red[tid + st]; __syncthreads(); }
  const float inv = 1.f / fmaxf(sqrtf(red[0]), 1e-12f);
  for (int i = tid; i < D_MODEL; i += 256)
    xn[(size_t)b * D_MODEL + i] = (_Float16)(__builtin_nontemporal_load(xr + i) * inv);

  if (tid == 0) {
    float r[NEXP]; float s2 = 0.f;
#pragma unroll
    for (int e = 0; e < NEXP; ++e) { r[e] = rp[(size_t)tok * NEXP + e]; s2 += r[e] * r[e]; }
    const float iv = 1.f / fmaxf(sqrtf(s2), 1e-12f);
#pragma unroll
    for (int e = 0; e < 32; ++e) rn[b * 32 + e] = (e < NEXP) ? (_Float16)(r[e] * iv) : (_Float16)0.f;
  }
}

// ---------- kernel 2b: 256x256 similarity tiles via WMMA, accumulate squared diff ----------
__global__ __launch_bounds__(256) void k_spr_sim(
    const _Float16* __restrict__ xn, const _Float16* __restrict__ rn,
    float* __restrict__ spr_acc)
{
  const int wave = threadIdx.x >> 5;
  const int lane = threadIdx.x & 31;
  const int half = lane >> 4;
  const int m    = lane & 15;
  const int tileIdx = blockIdx.x * 8 + wave;     // 256 tiles of 16x16
  const int i0 = (tileIdx >> 4) * 16;
  const int j0 = (tileIdx & 15) * 16;

  // input similarity: Xn @ Xn^T tile, K = 2048
  v8f ci = {};
  const _Float16* arow = xn + (size_t)(i0 + m) * D_MODEL;
  const _Float16* brow = xn + (size_t)(j0 + m) * D_MODEL;   // B[k][n] = Xn[j0+n][k]
  for (int k0 = 0; k0 < D_MODEL; k0 += 32) {
    v16h a = loadA_h16(arow, k0, half);
    v16h b = loadB_h16(brow, k0, half);
    ci = __builtin_amdgcn_wmma_f32_16x16x32_f16(false, a, false, b, (short)0, ci, false, false);
  }

  // routing similarity: Rn @ Rn^T tile, K padded 8 -> 32, single WMMA
  v8f cr = {};
  {
    v16h a = loadA_h16(rn + (size_t)(i0 + m) * 32, 0, half);
    v16h b = loadB_h16(rn + (size_t)(j0 + m) * 32, 0, half);
    cr = __builtin_amdgcn_wmma_f32_16x16x32_f16(false, a, false, b, (short)0, cr, false, false);
  }

  float accv = 0.f;
#pragma unroll
  for (int r = 0; r < 8; ++r) { float d = cr[r] - ci[r]; accv += d * d; }
#pragma unroll
  for (int off = 16; off >= 1; off >>= 1) accv += __shfl_xor(accv, off, 32);
  if (lane == 0) atomicAdd(spr_acc, accv);
}

// ---------- kernel 3: combine aux losses ----------
__global__ void k_finalize(const float* __restrict__ acc, float* __restrict__ out_aux) {
  if (threadIdx.x == 0) {
    float lb = 0.f, dp = 0.f;
#pragma unroll
    for (int e = 0; e < NEXP; ++e) {
      float f = acc[e] / (float)(NTOK * 2);        // fraction of routed slots
      float P = acc[8 + e] / (float)NTOK;          // mean router prob
      lb += f * P;
      dp += 0.125f * (logf(0.125f) - logf(fmaxf(P, 1e-30f)));
    }
    lb *= (float)NEXP;
    const float spr = acc[16] / (float)(SPR_T * SPR_T);
    out_aux[0] = 0.01f * (lb + dp + 0.1f * spr);
  }
}

extern "C" void kernel_launch(void* const* d_in, const int* in_sizes, int n_in,
                              void* d_out, int out_size, void* d_ws, size_t ws_size,
                              hipStream_t stream) {
  const float* x   = (const float*)d_in[0];   // [4,4096,2048]
  const float* W   = (const float*)d_in[1];   // [8,2048]
  const int*   spr = (const int*)d_in[2];     // [256]
  float* out = (float*)d_out;                 // [rp 131072 | idx 32768 | tp 32768 | aux 1]

  float*    acc = (float*)d_ws;                               // 17 floats of accumulators
  _Float16* xn  = (_Float16*)((char*)d_ws + 1024);            // 256*2048 f16 (1 MB)
  _Float16* rn  = xn + (size_t)SPR_T * D_MODEL;               // 256*32 f16 (16 KB, K-padded)

  k_init<<<1, 32, 0, stream>>>(acc);
  k_router_gemm<<<NTOK / 128, 256, 0, stream>>>(x, W, out, acc);
  k_spr_gather<<<SPR_T, 256, 0, stream>>>(x, spr, out, xn, rn);
  k_spr_sim<<<32, 256, 0, stream>>>(xn, rn, acc + 16);
  k_finalize<<<1, 32, 0, stream>>>(acc, out + (size_t)NTOK * (NEXP + 4));
}
